// BalSCL_SSL_BAL_16690242913044
// MI455X (gfx1250) — matrix operand order repaired and checked by reading, not compiled, not run on hardware
//
#include <hip/hip_runtime.h>
#include <hip/hip_bf16.h>

typedef __attribute__((ext_vector_type(16))) _Float16 v16h;
typedef __attribute__((ext_vector_type(8)))  _Float16 v8h;
typedef __attribute__((ext_vector_type(8)))  float    v8f;

#define B2      8192     // number of feature rows (anchors)
#define NC      100      // classes
#define DD      128      // feature dim
#define NTOT    8292     // B2 + NC columns
#define NPAD    8304     // 519 * 16 (padded columns)
#define NTILES  519      // column tiles of 16
#define NCHUNK  8        // column-parallel chunks
#define TPC     65       // tiles per chunk (8*65 >= 519)

// ---------------- prep kernels ----------------

__global__ void init_cls_kernel(int* __restrict__ clsc) {
    int c = threadIdx.x;
    if (c < 128) clsc[c] = 1;   // classes: 1 center each; slots >= NC: benign w=1 for pads
}

__global__ void prep_labels_kernel(const float* __restrict__ targets,
                                   int* __restrict__ labels,
                                   int* __restrict__ preds,
                                   int* __restrict__ clsc) {
    int j = blockIdx.x * blockDim.x + threadIdx.x;
    if (j >= NPAD) return;
    if (j < B2) {
        // one-hot argmax
        int lab = 0; float best = -1.0f;
        const float* row = targets + (size_t)j * NC;
        for (int c = 0; c < NC; ++c) {
            float t = row[c];
            if (t > best) { best = t; lab = c; }
        }
        labels[j] = lab;
        preds[j]  = lab;
        atomicAdd(&clsc[lab], 1);
    } else if (j < NTOT) {
        preds[j] = j - B2;   // center rows: identity one-hot
    } else {
        preds[j] = NC;       // pad: pred=100 never matches a label (isPos=false)
    }
}

__global__ void clsc_to_float_kernel(const int* __restrict__ clsc,
                                     float* __restrict__ clscF) {
    int c = threadIdx.x;
    if (c < 128) clscF[c] = (float)clsc[c];
}

__global__ void convert_f16_kernel(const float* __restrict__ feats,
                                   const float* __restrict__ centers,
                                   _Float16* __restrict__ FB) {
    int idx = blockIdx.x * blockDim.x + threadIdx.x;
    if (idx >= NPAD * DD) return;
    int row = idx / DD;
    float v;
    if      (row < B2)   v = feats[idx];
    else if (row < NTOT) v = centers[idx - B2 * DD];
    else                 v = 0.0f;     // zero pad rows -> logits 0, masked out of S
    FB[idx] = (_Float16)v;
}

// ---------------- fused GEMM + softmax-reduction kernel ----------------

__device__ __forceinline__ v16h load_fragA(const _Float16* rowptr, int kbase, int h) {
    // ISA 16-bit A layout: lane half h holds K = {kbase+8h..+7} U {kbase+16+8h..+7}
    v8h lo = *(const v8h*)(rowptr + kbase + 8 * h);
    v8h hi = *(const v8h*)(rowptr + kbase + 16 + 8 * h);
    return __builtin_shufflevector(lo, hi, 0,1,2,3,4,5,6,7,8,9,10,11,12,13,14,15);
}

__device__ __forceinline__ v16h load_fragB(const _Float16* rowptr, int kbase, int h) {
    // ISA 16-bit B layout: lane half h holds K = {kbase+8h..+7} U {kbase+16+8h..+7}
    v8h lo = *(const v8h*)(rowptr + kbase + 8 * h);
    v8h hi = *(const v8h*)(rowptr + kbase + 16 + 8 * h);
    return __builtin_shufflevector(lo, hi, 0,1,2,3,4,5,6,7,8,9,10,11,12,13,14,15);
}

__global__ __launch_bounds__(256)
void bal_scl_main_kernel(const _Float16* __restrict__ FB,
                         const int* __restrict__ preds,
                         const int* __restrict__ labels,
                         const float* __restrict__ clscF,
                         float* __restrict__ Sp,
                         float* __restrict__ Pp) {
    const int lane    = threadIdx.x & 31;
    const int wave    = threadIdx.x >> 5;
    const int rowTile = blockIdx.x * 8 + wave;        // 64 blocks * 8 waves = 512 tiles
    const int rowBase = rowTile * 16;
    const int chunk   = blockIdx.y;                    // 8 column chunks
    const int li      = lane & 15;
    const int h       = lane >> 4;

    // Resident A fragments: 16 rows x K=128 (4 K-fragments of 32)
    const _Float16* arow = FB + (size_t)(rowBase + li) * DD;
    v16h a0 = load_fragA(arow,  0, h);
    v16h a1 = load_fragA(arow, 32, h);
    v16h a2 = load_fragA(arow, 64, h);
    v16h a3 = load_fragA(arow, 96, h);

    // Labels for the 8 rows this lane's C-fragment covers (M = 8h + r)
    int lab[8];
    #pragma unroll
    for (int r = 0; r < 8; ++r) lab[r] = labels[rowBase + 8 * h + r];

    float Sacc[8], Pacc[8];
    #pragma unroll
    for (int r = 0; r < 8; ++r) { Sacc[r] = 0.0f; Pacc[r] = 0.0f; }

    const int t0 = chunk * TPC;
    const int t1 = (t0 + TPC < NTILES) ? (t0 + TPC) : NTILES;

    for (int t = t0; t < t1; ++t) {
        const int colBase = t * 16;
        const int col     = colBase + li;              // max 8303 < NPAD, in-bounds
        const _Float16* brow = FB + (size_t)col * DD;

        v16h b0 = load_fragB(brow,  0, h);
        v16h b1 = load_fragB(brow, 32, h);
        v16h b2 = load_fragB(brow, 64, h);
        v16h b3 = load_fragB(brow, 96, h);

        v8f c = {};
        c = __builtin_amdgcn_wmma_f32_16x16x32_f16(false, a0, false, b0, (short)0, c, false, false);
        c = __builtin_amdgcn_wmma_f32_16x16x32_f16(false, a1, false, b1, (short)0, c, false, false);
        c = __builtin_amdgcn_wmma_f32_16x16x32_f16(false, a2, false, b2, (short)0, c, false, false);
        c = __builtin_amdgcn_wmma_f32_16x16x32_f16(false, a3, false, b3, (short)0, c, false, false);

        const float vm   = (col < NTOT) ? 1.0f : 0.0f; // one mask per tile (gates S only)
        const int   pred = preds[col];                 // pad reads pred=NC (never a label)
        const float cc   = clscF[pred];

        // Self column included as a regular positive; corrected in finalize.
        #pragma unroll
        for (int r = 0; r < 8; ++r) {
            const float earg   = __builtin_fmaf(c[r], 10.0f, -10.0f); // l - SHIFT
            const float isPosF = (pred == lab[r]) ? 1.0f : 0.0f;
            const float w      = cc - isPosF;                         // >= 1 always
            const float e      = __expf(earg) * __builtin_amdgcn_rcpf(w);
            Sacc[r] = __builtin_fmaf(e, vm, Sacc[r]);
            Pacc[r] = __builtin_fmaf(earg, isPosF, Pacc[r]);          // pads: isPosF=0
        }
    }

    // Reduce over the 16 N-lanes of each half-wave (fixed order -> deterministic)
    #pragma unroll
    for (int r = 0; r < 8; ++r) {
        float s = Sacc[r], p = Pacc[r];
        #pragma unroll
        for (int m = 8; m >= 1; m >>= 1) {
            s += __shfl_xor(s, m, 16);
            p += __shfl_xor(p, m, 16);
        }
        Sacc[r] = s; Pacc[r] = p;
    }
    if (li == 0) {
        #pragma unroll
        for (int r = 0; r < 8; ++r) {
            const int grow = rowBase + 8 * h + r;
            Sp[grow * NCHUNK + chunk] = Sacc[r];
            Pp[grow * NCHUNK + chunk] = Pacc[r];
        }
    }
}

// ---------------- final scalar reduction ----------------

__global__ __launch_bounds__(256)
void finalize_kernel(const float* __restrict__ Sp, const float* __restrict__ Pp,
                     const _Float16* __restrict__ FB,
                     const int* __restrict__ labels, const int* __restrict__ clsc,
                     const float* __restrict__ conf, float* __restrict__ out) {
    __shared__ float sn[256];
    __shared__ float sd[256];
    float num = 0.0f, den = 0.0f;
    for (int i = threadIdx.x; i < B2; i += 256) {
        float S = 0.0f, P = 0.0f;
        #pragma unroll
        for (int k = 0; k < NCHUNK; ++k) { S += Sp[i * NCHUNK + k]; P += Pp[i * NCHUNK + k]; }
        const float cf  = conf[i];
        const int   L   = labels[i];
        const float cnt = (float)(clsc[L] - 1);       // >= 1 (row i itself was counted)
        // Recompute the self logit from the f16 data
        const _Float16* frow = FB + (size_t)i * DD;
        float sdot = 0.0f;
        for (int k = 0; k < DD; ++k) {
            const float v = (float)frow[k];
            sdot = __builtin_fmaf(v, v, sdot);
        }
        const float earg_self = __builtin_fmaf(sdot, 10.0f, -10.0f);  // l_self - SHIFT
        const float rc     = 1.0f / cnt;
        const float S_true = S - __expf(earg_self) * rc;  // self had w = cnt in the loop
        const float Pexcl  = P - earg_self;               // shifted positive sum, no self
        const float v      = logf(S_true) - Pexcl * rc;   // SHIFT cancels exactly
        num += cf * v;
        den += cf;
    }
    sn[threadIdx.x] = num;
    sd[threadIdx.x] = den;
    __syncthreads();
    for (int s = 128; s > 0; s >>= 1) {
        if (threadIdx.x < s) {
            sn[threadIdx.x] += sn[threadIdx.x + s];
            sd[threadIdx.x] += sd[threadIdx.x + s];
        }
        __syncthreads();
    }
    if (threadIdx.x == 0) out[0] = sn[0] / sd[0];
}

// ---------------- launch ----------------

extern "C" void kernel_launch(void* const* d_in, const int* in_sizes, int n_in,
                              void* d_out, int out_size, void* d_ws, size_t ws_size,
                              hipStream_t stream) {
    const float* centers  = (const float*)d_in[0];   // [100,128]
    const float* features = (const float*)d_in[1];   // [8192,128]
    const float* targets  = (const float*)d_in[2];   // [8192,100]
    // d_in[3] = cls_num_list (all ones, only shape used) -> unused
    const float* conf     = (const float*)d_in[4];   // [8192]
    float* out = (float*)d_out;

    // Workspace carve (256B aligned), ~2.7 MB total
    uint8_t* base = (uint8_t*)d_ws;
    size_t off = 0;
    auto carve = [&](size_t bytes) -> void* {
        void* p = base + off;
        off += (bytes + 255) & ~(size_t)255;
        return p;
    };
    _Float16* FB     = (_Float16*)carve((size_t)NPAD * DD * sizeof(_Float16));
    float*    Sp     = (float*)   carve((size_t)B2 * NCHUNK * sizeof(float));
    float*    Pp     = (float*)   carve((size_t)B2 * NCHUNK * sizeof(float));
    int*      labels = (int*)     carve((size_t)B2 * sizeof(int));
    int*      preds  = (int*)     carve((size_t)NPAD * sizeof(int));
    int*      clsc   = (int*)     carve(128 * sizeof(int));
    float*    clscF  = (float*)   carve(128 * sizeof(float));
    (void)ws_size; (void)in_sizes; (void)n_in; (void)out_size;

    init_cls_kernel<<<1, 128, 0, stream>>>(clsc);
    prep_labels_kernel<<<(NPAD + 255) / 256, 256, 0, stream>>>(targets, labels, preds, clsc);
    clsc_to_float_kernel<<<1, 128, 0, stream>>>(clsc, clscF);
    convert_f16_kernel<<<(NPAD * DD + 255) / 256, 256, 0, stream>>>(features, centers, FB);

    dim3 grid(64, NCHUNK);   // 512 row tiles as 64 blocks x 8 waves; 8 column chunks
    bal_scl_main_kernel<<<grid, 256, 0, stream>>>(FB, preds, labels, clscF, Sp, Pp);

    finalize_kernel<<<1, 256, 0, stream>>>(Sp, Pp, FB, labels, clsc, conf, out);
}
